// GINNet_28707561406530
// MI455X (gfx1250) — matrix operand order repaired and checked by reading, not compiled, not run on hardware
//
#include <hip/hip_runtime.h>
#include <hip/hip_bf16.h>
#include <math.h>
#include <stdint.h>

typedef __attribute__((ext_vector_type(16))) _Float16 v16h;
typedef __attribute__((ext_vector_type(8)))  _Float16 v8h;
typedef __attribute__((ext_vector_type(8)))  float    v8f;

#define N_NODES 150000
#define N_EDGES 600000
#define BGR     512
#define FXD     334
#define GDIM    32
#define LMUT    13132
// conv geometry: L-7 conv, /3 pool
#define L1P 4375
#define L2C 4368
#define L2P 1456
#define L3C 1449
#define L3P 483
#define FLAT_XT 61824  // 128*483
#define XT_SPLITS 12   // 1932 K-chunks / 12 = 161 exactly

__device__ __forceinline__ v16h cat8(v8h lo, v8h hi) {
  return __builtin_shufflevector(lo, hi, 0,1,2,3,4,5,6,7,8,9,10,11,12,13,14,15);
}

// CDNA5 async global->LDS copy (ASYNCcnt path, CDNA5 ISA)
__device__ __forceinline__ void async_ld_b32(unsigned lds_off, const _Float16* g) {
  asm volatile("global_load_async_to_lds_b32 %0, %1, off"
               :: "v"(lds_off), "v"(g) : "memory");
}
__device__ __forceinline__ void wait_async0() {
  asm volatile("s_wait_asynccnt 0" ::: "memory");
}

// ---------------------------------------------------------------------------
// WMMA GEMM: C[M,N] = act(A[M,K] * BT^T + bias), BT is [N][Kpad] f16 transposed
// and zero-padded. A is f32 (guarded scalar loads) or f16 (two v8h loads,
// requires K == Kpad). One wave per 16x16 tile; grid (M/16, N/16[, splits]).
// SPLITK: grid.z slices the K-chunks; partials accumulated with f32 atomics
// into a pre-zeroed C (no relu/f16 out in that mode; bias added by slice 0).
// ---------------------------------------------------------------------------
template<bool A_F32, bool RELU, bool OUT_F16, bool HAS_BIAS, bool SPLITK>
__global__ void wmma_gemm_kernel(const void* __restrict__ Aptr,
                                 const _Float16* __restrict__ BT,
                                 const float* __restrict__ bias,
                                 void* __restrict__ Cptr,
                                 int K, int lda, int ldbt, int ldc, int cps) {
  const int l  = threadIdx.x & 31;
  const int m0 = blockIdx.x * 16;
  const int n0 = blockIdx.y * 16;
  const int r  = l & 15;
  const int kh = (l >> 4) & 1;
  v8f acc = {};
  const int totalChunks = ldbt >> 5;
  int cStart = 0, cEnd = totalChunks;
  if (SPLITK) {
    cStart = blockIdx.z * cps;
    cEnd   = min(cStart + cps, totalChunks);
  }
  const _Float16* brow = BT + (size_t)(n0 + r) * ldbt + kh * 8;
  if (A_F32) {
    const float* Af = (const float*)Aptr;
    const size_t arow = (size_t)(m0 + r) * (size_t)lda;
    for (int c = cStart; c < cEnd; ++c) {
      const int k0 = c << 5;
      if (c + 1 < cEnd) __builtin_prefetch(brow + k0 + 32, 0, 1);
      v16h a;
#pragma unroll
      for (int i = 0; i < 8; ++i) {
        const int kb = (i < 4 ? 2 * i : 16 + 2 * (i - 4)) + kh * 8;
        const int ka = k0 + kb;
        a[2 * i]     = (ka < K)     ? (_Float16)Af[arow + ka]     : (_Float16)0.f;
        a[2 * i + 1] = (ka + 1 < K) ? (_Float16)Af[arow + ka + 1] : (_Float16)0.f;
      }
      v16h b = cat8(*(const v8h*)(brow + k0), *(const v8h*)(brow + k0 + 16));
      acc = __builtin_amdgcn_wmma_f32_16x16x32_f16(false, a, false, b, (short)0,
                                                   acc, false, false);
    }
  } else {
    const _Float16* Ah = (const _Float16*)Aptr + (size_t)(m0 + r) * lda + kh * 8;
    for (int c = cStart; c < cEnd; ++c) {
      const int k0 = c << 5;
      if (c + 1 < cEnd) __builtin_prefetch(brow + k0 + 32, 0, 1);
      v16h a = cat8(*(const v8h*)(Ah + k0), *(const v8h*)(Ah + k0 + 16));
      v16h b = cat8(*(const v8h*)(brow + k0), *(const v8h*)(brow + k0 + 16));
      acc = __builtin_amdgcn_wmma_f32_16x16x32_f16(false, a, false, b, (short)0,
                                                   acc, false, false);
    }
  }
#pragma unroll
  for (int i = 0; i < 8; ++i) {
    const int m = i + kh * 8;
    float v = acc[i];
    const size_t idx = (size_t)(m0 + m) * ldc + (n0 + r);
    if (SPLITK) {
      if (HAS_BIAS && blockIdx.z == 0) v += bias[n0 + r];
      atomicAdd(&((float*)Cptr)[idx], v);
    } else {
      if (HAS_BIAS) v += bias[n0 + r];
      if (RELU) v = v > 0.f ? v : 0.f;
      if (OUT_F16) ((_Float16*)Cptr)[idx] = (_Float16)v;
      else         ((float*)Cptr)[idx]    = v;
    }
  }
}

// ---------------------------------------------------------------------------
// Fused conv1d(k=8)+bias+relu+maxpool3 implicit GEMM. Block = (batch, 16-wide
// pooled tile), NW = OC/16 waves, each owning one oc tile. Input patch staged
// to LDS with async b32 loads; odd ic rows skewed by one half so every b32 is
// 4B-aligned (row pitch 58 halves). Three pool-phase-strided N tiles make the
// pool max lane-local.
// ---------------------------------------------------------------------------
template<int IC, int OC, int LIN, int LCONV, int LPOOL, int NW>
__global__ void conv_wmma_pool_kernel(const _Float16* __restrict__ in,   // [B][IC][LIN]
                                      const _Float16* __restrict__ W,    // [OC][IC*8]
                                      const float* __restrict__ bias,
                                      _Float16* __restrict__ out) {      // [B][OC][LPOOL]
  constexpr int ROWP = 58;
  constexpr int Kd = IC * 8;
  constexpr int NTH = 32 * NW;
  constexpr int TOTD = IC * 28;                 // dwords staged (56 halves/row)
  static_assert(TOTD % NTH == 0, "stage divides");
  __shared__ __align__(16) _Float16 smem[IC * ROWP];
  const int b   = blockIdx.x;
  const int pp0 = blockIdx.y * 16;
  const int tid = threadIdx.x;
  const _Float16* inB = in + (size_t)b * IC * LIN + 3 * pp0;
  const unsigned sbase = (unsigned)(size_t)smem;
#pragma unroll
  for (int it = 0; it < TOTD / NTH; ++it) {
    const int e  = tid + it * NTH;
    const int ic = e / 28, w = e % 28;
    // odd rows start 2-mod-4; shift by one half to keep b32 aligned
    const _Float16* g = inB + (size_t)ic * LIN + 2 * w - (ic & 1);
    async_ld_b32(sbase + (unsigned)(ic * ROWP + 2 * w) * 2u, g);
  }
  wait_async0();
  __syncthreads();

  const int l   = tid & 31;
  const int r   = l & 15;
  const int kh  = (l >> 4) & 1;
  const int oc0 = (tid >> 5) * 16;
  const v8f zero = {};
  v8f acc[3] = {zero, zero, zero};
  const _Float16* wrow = W + (size_t)(oc0 + r) * Kd + kh * 16;
  for (int c = 0; c < Kd / 32; ++c) {
    const int k0 = c << 5;
    v16h a = cat8(*(const v8h*)(wrow + k0), *(const v8h*)(wrow + k0 + 8));
    // chunk covers ic = 4c + 2kh (even, skew 0) and 4c + 2kh + 1 (odd, skew 1)
    const _Float16* ra = &smem[(4 * c + 2 * kh) * ROWP];
#pragma unroll
    for (int j = 0; j < 3; ++j) {
      const int pl = 3 * r + j;
      v16h bf;
#pragma unroll
      for (int q = 0; q < 8; ++q) {
        bf[q]     = ra[pl + q];                  // ic_a, kk = q
        bf[8 + q] = ra[ROWP + 1 + pl + q];       // ic_a+1 (skewed), kk = q
      }
      acc[j] = __builtin_amdgcn_wmma_f32_16x16x32_f16(false, a, false, bf,
                                                      (short)0, acc[j], false, false);
    }
  }
#pragma unroll
  for (int i = 0; i < 8; ++i) {
    const int m = i + kh * 8;
    if (pp0 + r < LPOOL) {
      float v = fmaxf(fmaxf(acc[0][i], acc[1][i]), acc[2][i]) + bias[oc0 + m];
      out[((size_t)b * OC + oc0 + m) * LPOOL + pp0 + r] = (_Float16)(v > 0.f ? v : 0.f);
    }
  }
}

// ----------------------- scalar helper kernels -----------------------------
__global__ void f32_to_f16_kernel(const float* __restrict__ s,
                                  _Float16* __restrict__ d, int n) {
  int i = blockIdx.x * blockDim.x + threadIdx.x;
  if (i < n) d[i] = (_Float16)s[i];
}

// dst[n][k] = src[k][n] (f16, zero-padded to Kpad)
__global__ void transpose_f16_kernel(const float* __restrict__ src,
                                     _Float16* __restrict__ dst,
                                     int K, int N, int Kpad) {
  int idx = blockIdx.x * blockDim.x + threadIdx.x;
  if (idx >= N * Kpad) return;
  const int n = idx / Kpad, k = idx % Kpad;
  dst[idx] = (k < K) ? (_Float16)src[(size_t)k * N + n] : (_Float16)0.f;
}

__global__ void conv1_kernel(const float* __restrict__ x,     // [512][13132]
                             const float* __restrict__ w,     // [32][8]
                             const float* __restrict__ bias,
                             _Float16* __restrict__ out) {    // [512][32][4375]
  int idx = blockIdx.x * blockDim.x + threadIdx.x;
  if (idx >= BGR * 32 * L1P) return;
  const int p  = idx % L1P;
  const int oc = (idx / L1P) % 32;
  const int b  = idx / (L1P * 32);
  const float* xb = x + (size_t)b * LMUT;
  float mx = -1e30f;
#pragma unroll
  for (int j = 0; j < 3; ++j) {
    float s = bias[oc];
    const int base = 3 * p + j;
#pragma unroll
    for (int k = 0; k < 8; ++k) s += xb[base + k] * w[oc * 8 + k];
    mx = fmaxf(mx, s);
  }
  out[idx] = (_Float16)fmaxf(mx, 0.f);
}

__global__ void scatter_add_kernel(const float* __restrict__ t,
                                   const int* __restrict__ ei,
                                   float* __restrict__ agg) {
  int idx = blockIdx.x * blockDim.x + threadIdx.x;
  if (idx >= N_EDGES * GDIM) return;
  const int e = idx >> 5, f = idx & 31;
  const int s = ei[e], d = ei[N_EDGES + e];
  atomicAdd(&agg[(size_t)d * GDIM + f], t[(size_t)s * GDIM + f]);
}

// z1 = relu(t + agg + b1), float4-vectorized (stride 32 -> bias index idx&7)
__global__ void z1_kernel(const float4* __restrict__ t, const float4* __restrict__ agg,
                          const float4* __restrict__ b1, float4* __restrict__ z1) {
  int idx = blockIdx.x * blockDim.x + threadIdx.x;
  if (idx >= N_NODES * GDIM / 4) return;
  const float4 a = t[idx], g = agg[idx], bb = b1[idx & 7];
  float4 o;
  o.x = fmaxf(a.x + g.x + bb.x, 0.f);
  o.y = fmaxf(a.y + g.y + bb.y, 0.f);
  o.z = fmaxf(a.z + g.z + bb.z, 0.f);
  o.w = fmaxf(a.w + g.w + bb.w, 0.f);
  z1[idx] = o;
}

__global__ void bn_stats_kernel(const float* __restrict__ z, float* __restrict__ st) {
  __shared__ float ssum[256], ssq[256];
  const int tid = threadIdx.x;
  const int c = tid & 31, r = tid >> 5;
  float s = 0.f, q = 0.f;
  for (int i = blockIdx.x * 8 + r; i < N_NODES; i += gridDim.x * 8) {
    float v = z[(size_t)i * GDIM + c];
    s += v; q += v * v;
  }
  ssum[tid] = s; ssq[tid] = q;
  __syncthreads();
  if (r == 0) {
#pragma unroll
    for (int j = 1; j < 8; ++j) { s += ssum[j * 32 + c]; q += ssq[j * 32 + c]; }
    atomicAdd(&st[c], s);
    atomicAdd(&st[32 + c], q);
  }
}

// h = (z-mu)*rsqrt(var+eps)*gamma + beta, float4-vectorized
__global__ void bn_apply_kernel(const float4* __restrict__ z, const float4* __restrict__ st,
                                const float4* __restrict__ gamma, const float4* __restrict__ beta,
                                float4* __restrict__ h) {
  int idx = blockIdx.x * blockDim.x + threadIdx.x;
  if (idx >= N_NODES * GDIM / 4) return;
  const int c4 = idx & 7;
  const float inv = 1.f / (float)N_NODES;
  const float4 s1 = st[c4], s2 = st[8 + c4], gm = gamma[c4], bt = beta[c4];
  const float4 zv = z[idx];
  float4 o;
  float mu, var;
  mu = s1.x * inv; var = s2.x * inv - mu * mu; o.x = (zv.x - mu) * rsqrtf(var + 1e-5f) * gm.x + bt.x;
  mu = s1.y * inv; var = s2.y * inv - mu * mu; o.y = (zv.y - mu) * rsqrtf(var + 1e-5f) * gm.y + bt.y;
  mu = s1.z * inv; var = s2.z * inv - mu * mu; o.z = (zv.z - mu) * rsqrtf(var + 1e-5f) * gm.z + bt.z;
  mu = s1.w * inv; var = s2.w * inv - mu * mu; o.w = (zv.w - mu) * rsqrtf(var + 1e-5f) * gm.w + bt.w;
  h[idx] = o;
}

__global__ void pool_kernel(const float* __restrict__ h, const int* __restrict__ batch,
                            float* __restrict__ xdp) {
  int idx = blockIdx.x * blockDim.x + threadIdx.x;
  if (idx >= N_NODES * GDIM) return;
  const int i = idx >> 5, f = idx & 31;
  atomicAdd(&xdp[(size_t)batch[i] * GDIM + f], h[idx]);
}

__global__ void concat_kernel(const float* __restrict__ xd, const float* __restrict__ xt,
                              _Float16* __restrict__ xc) {
  int idx = blockIdx.x * blockDim.x + threadIdx.x;
  if (idx >= BGR * 256) return;
  const int b = idx >> 8, f = idx & 255;
  const float v = (f < 128) ? xd[b * 128 + f] : xt[b * 128 + f - 128];
  xc[idx] = (_Float16)v;
}

__global__ void out_kernel(const float* __restrict__ f2, const float* __restrict__ w,
                           const float* __restrict__ b0, float* __restrict__ out) {
  int b = blockIdx.x * blockDim.x + threadIdx.x;
  if (b >= BGR) return;
  float s = b0[0];
#pragma unroll 8
  for (int k = 0; k < 128; ++k) s += f2[b * 128 + k] * w[k];
  out[b] = 1.f / (1.f + expf(-s));
}

// ---------------------------------------------------------------------------
extern "C" void kernel_launch(void* const* d_in, const int* in_sizes, int n_in,
                              void* d_out, int out_size, void* d_ws, size_t ws_size,
                              hipStream_t stream) {
  const float* x     = (const float*)d_in[0];
  const int*   ei    = (const int*)d_in[1];
  const int*   batch = (const int*)d_in[2];
  const float* xmut  = (const float*)d_in[3];
  // d_in[4] edge_feat: unused by the reference

  const float *gW1[5], *gb1[5], *gW2[5], *gb2[5], *gGm[5], *gBt[5];
  for (int l = 0; l < 5; ++l) {
    gW1[l] = (const float*)d_in[5 + 6 * l];  gb1[l] = (const float*)d_in[6 + 6 * l];
    gW2[l] = (const float*)d_in[7 + 6 * l];  gb2[l] = (const float*)d_in[8 + 6 * l];
    gGm[l] = (const float*)d_in[9 + 6 * l];  gBt[l] = (const float*)d_in[10 + 6 * l];
  }
  const float* fc1xdW = (const float*)d_in[35];
  const float* fc1xdB = (const float*)d_in[36];
  const float* cw1 = (const float*)d_in[37];
  const float* cb1 = (const float*)d_in[38];
  const float* cw2 = (const float*)d_in[39];
  const float* cb2 = (const float*)d_in[40];
  const float* cw3 = (const float*)d_in[41];
  const float* cb3 = (const float*)d_in[42];
  const float* fxtW = (const float*)d_in[43];
  const float* fxtB = (const float*)d_in[44];
  const float* fc1W = (const float*)d_in[45];
  const float* fc1B = (const float*)d_in[46];
  const float* fc2W = (const float*)d_in[47];
  const float* fc2B = (const float*)d_in[48];
  const float* outW = (const float*)d_in[49];
  const float* outB = (const float*)d_in[50];

  // --- workspace bump allocator (~416 MB) ---
  char* base = (char*)d_ws;
  size_t off = 0;
  auto alloc = [&](size_t bytes) -> char* {
    char* p = base + off;
    off = (off + bytes + 255) & ~(size_t)255;
    return p;
  };
  float* t     = (float*)alloc((size_t)N_NODES * GDIM * 4);
  float* agg   = (float*)alloc((size_t)N_NODES * GDIM * 4);
  float* z1b   = (float*)alloc((size_t)N_NODES * GDIM * 4);
  float* hA    = (float*)alloc((size_t)N_NODES * GDIM * 4);
  float* hB    = (float*)alloc((size_t)N_NODES * GDIM * 4);
  float* st    = (float*)alloc(64 * 4);
  float* xdp   = (float*)alloc((size_t)BGR * GDIM * 4);
  float* xd    = (float*)alloc((size_t)BGR * 128 * 4);
  float* xt    = (float*)alloc((size_t)BGR * 128 * 4);
  float* f2o   = (float*)alloc((size_t)BGR * 128 * 4);
  _Float16* c1 = (_Float16*)alloc((size_t)BGR * 32 * L1P * 2);
  _Float16* c2 = (_Float16*)alloc((size_t)BGR * 64 * L2P * 2);
  _Float16* c3 = (_Float16*)alloc((size_t)BGR * 128 * L3P * 2);   // keep after c2 (stage overrun pad)
  _Float16* xc16 = (_Float16*)alloc((size_t)BGR * 256 * 2);
  _Float16* f116 = (_Float16*)alloc((size_t)BGR * 1024 * 2);
  // transposed+padded [N][Kpad] f16 weights for GEMM B operands
  _Float16* gW1t[5]; _Float16* gW2t[5];
  gW1t[0] = (_Float16*)alloc((size_t)GDIM * 352 * 2);
  for (int l = 1; l < 5; ++l) gW1t[l] = (_Float16*)alloc(GDIM * GDIM * 2);
  for (int l = 0; l < 5; ++l) gW2t[l] = (_Float16*)alloc(GDIM * GDIM * 2);
  _Float16* fc1xdWt = (_Float16*)alloc(128 * GDIM * 2);
  _Float16* fxtWt   = (_Float16*)alloc((size_t)128 * FLAT_XT * 2);
  _Float16* fc1Wt   = (_Float16*)alloc(1024 * 256 * 2);
  _Float16* fc2Wt   = (_Float16*)alloc(128 * 1024 * 2);
  // conv weights stay row-major [OC][IC*8] (A operand)
  _Float16* w2h = (_Float16*)alloc(64 * 256 * 2);
  _Float16* w3h = (_Float16*)alloc(128 * 512 * 2);

  auto cvt = [&](const float* s, _Float16* d, int n) {
    f32_to_f16_kernel<<<(n + 255) / 256, 256, 0, stream>>>(s, d, n);
  };
  auto tp = [&](const float* s, _Float16* d, int K, int N, int Kpad) {
    int n = N * Kpad;
    transpose_f16_kernel<<<(n + 255) / 256, 256, 0, stream>>>(s, d, K, N, Kpad);
  };
  tp(gW1[0], gW1t[0], FXD, GDIM, 352);
  for (int l = 1; l < 5; ++l) tp(gW1[l], gW1t[l], GDIM, GDIM, GDIM);
  for (int l = 0; l < 5; ++l) tp(gW2[l], gW2t[l], GDIM, GDIM, GDIM);
  tp(fc1xdW, fc1xdWt, GDIM, 128, GDIM);
  tp(fxtW, fxtWt, FLAT_XT, 128, FLAT_XT);
  tp(fc1W, fc1Wt, 256, 1024, 256);
  tp(fc2W, fc2Wt, 1024, 128, 1024);
  cvt(cw2, w2h, 64 * 256);
  cvt(cw3, w3h, 128 * 512);

  // ---------------- CNN branch ----------------
  conv1_kernel<<<(BGR * 32 * L1P + 255) / 256, 256, 0, stream>>>(xmut, cw1, cb1, c1);
  conv_wmma_pool_kernel<32, 64, L1P, L2C, L2P, 4>
      <<<dim3(BGR, L2P / 16), 128, 0, stream>>>(c1, w2h, cb2, c2);
  conv_wmma_pool_kernel<64, 128, L2P, L3C, L3P, 8>
      <<<dim3(BGR, (L3P + 15) / 16), 256, 0, stream>>>(c2, w3h, cb3, c3);
  // xt = flatten(c3) @ fc1_xt_W + b, split-K x12 for occupancy (1932/12=161)
  hipMemsetAsync(xt, 0, (size_t)BGR * 128 * 4, stream);
  wmma_gemm_kernel<false, false, false, true, true>
      <<<dim3(BGR / 16, 128 / 16, XT_SPLITS), 32, 0, stream>>>(
          c3, fxtWt, fxtB, xt, FLAT_XT, FLAT_XT, FLAT_XT, 128,
          (FLAT_XT / 32) / XT_SPLITS);

  // ---------------- GIN branch ----------------
  const int ew  = (N_NODES * GDIM + 255) / 256;
  const int ew4 = (N_NODES * GDIM / 4 + 255) / 256;
  float* hbuf[2] = {hA, hB};
  const void* hprev = (const void*)x;
  for (int l = 0; l < 5; ++l) {
    const int K = (l == 0) ? FXD : GDIM;
    const int Kp = (l == 0) ? 352 : GDIM;
    // t = h @ W1 (linearity: aggregate AFTER projection -> 10x less traffic)
    wmma_gemm_kernel<true, false, false, false, false>
        <<<dim3(N_NODES / 16, GDIM / 16), 32, 0, stream>>>(hprev, gW1t[l], nullptr,
                                                           t, K, K, Kp, GDIM, 0);
    hipMemsetAsync(agg, 0, (size_t)N_NODES * GDIM * 4, stream);
    scatter_add_kernel<<<(N_EDGES * GDIM + 255) / 256, 256, 0, stream>>>(t, ei, agg);
    z1_kernel<<<ew4, 256, 0, stream>>>((const float4*)t, (const float4*)agg,
                                       (const float4*)gb1[l], (float4*)z1b);
    wmma_gemm_kernel<true, true, false, true, false>
        <<<dim3(N_NODES / 16, GDIM / 16), 32, 0, stream>>>(z1b, gW2t[l], gb2[l],
                                                           t, GDIM, GDIM, GDIM, GDIM, 0);
    hipMemsetAsync(st, 0, 64 * 4, stream);
    bn_stats_kernel<<<120, 256, 0, stream>>>(t, st);
    float* hout = hbuf[l & 1];
    bn_apply_kernel<<<ew4, 256, 0, stream>>>((const float4*)t, (const float4*)st,
                                             (const float4*)gGm[l], (const float4*)gBt[l],
                                             (float4*)hout);
    hprev = (const void*)hout;
  }
  hipMemsetAsync(xdp, 0, (size_t)BGR * GDIM * 4, stream);
  pool_kernel<<<ew, 256, 0, stream>>>((const float*)hprev, batch, xdp);
  wmma_gemm_kernel<true, true, false, true, false>
      <<<dim3(BGR / 16, 128 / 16), 32, 0, stream>>>(xdp, fc1xdWt, fc1xdB, xd,
                                                    GDIM, GDIM, GDIM, 128, 0);

  // ---------------- dense tail ----------------
  concat_kernel<<<(BGR * 256 + 255) / 256, 256, 0, stream>>>(xd, xt, xc16);
  wmma_gemm_kernel<false, true, true, true, false>
      <<<dim3(BGR / 16, 1024 / 16), 32, 0, stream>>>(xc16, fc1Wt, fc1B, f116,
                                                     256, 256, 256, 1024, 0);
  wmma_gemm_kernel<false, true, false, true, false>
      <<<dim3(BGR / 16, 128 / 16), 32, 0, stream>>>(f116, fc2Wt, fc2B, f2o,
                                                    1024, 1024, 1024, 128, 0);
  out_kernel<<<(BGR + 255) / 256, 256, 0, stream>>>(f2o, outW, outB, (float*)d_out);
}
